// CausalSelfAttention_68822555951130
// MI455X (gfx1250) — matrix operand order repaired and checked by reading, not compiled, not run on hardware
//
#include <hip/hip_runtime.h>
#include <hip/hip_bf16.h>

// ---------------------------------------------------------------------------
// Causal self-attention, MI455X (gfx1250, wave32, WMMA).
//   qkv = x @ Wqkv^T ; flash-attention per head ; out = attn @ Wout^T
// All matmuls on v_wmma_f32_16x16x32_bf16 (bf16 operands, f32 accum).
// Softmax statistics in f32; row-sum folded into an extra WMMA (ones column).
// Q pre-scaled by 1/sqrt(64)=0.125 (exact in bf16). Scores never materialized.
// ---------------------------------------------------------------------------

typedef __attribute__((ext_vector_type(16))) __bf16 bf16x16;
typedef __attribute__((ext_vector_type(8)))  __bf16 bf16x8;
typedef __attribute__((ext_vector_type(8)))  float  f32x8;

#define T_SEQ 4096
#define C_DIM 768
#define NH    12
#define HD    64
#define QKV_N (3 * C_DIM)

#define WMMA_BF16(a, b, c) \
  __builtin_amdgcn_wmma_f32_16x16x32_bf16(false, (a), false, (b), (short)0, (c), false, false)

// A-operand tile loader: 16x32 bf16 from row-major [rows][ld].
//   lanes 0-15 : row=lane,    K in {0..7, 16..23}
//   lanes 16-31: row=lane-16, K in {8..15, 24..31}
__device__ __forceinline__ bf16x16 load_a16x32(const __bf16* __restrict__ base,
                                               int ld, int row0, int k0) {
  int lane = threadIdx.x & 31;
  int m    = lane & 15;
  int half = lane >> 4;
  const __bf16* p = base + (row0 + m) * ld + k0 + half * 8;
  bf16x8 lo = *(const bf16x8*)p;
  bf16x8 hi = *(const bf16x8*)(p + 16);
  bf16x16 r;
#pragma unroll
  for (int i = 0; i < 8; ++i) { r[i] = lo[i]; r[i + 8] = hi[i]; }
  return r;
}

// B-operand tile loader for logical B[k][n] = src[n][k] (src row-major, K contiguous).
//   lanes 0-15 : col=lane,    K in {0..7, 16..23}
//   lanes 16-31: col=lane-16, K in {8..15, 24..31}
__device__ __forceinline__ bf16x16 load_bT16x32(const __bf16* __restrict__ base,
                                                int ld, int col0, int k0) {
  int lane = threadIdx.x & 31;
  int n    = lane & 15;
  int half = lane >> 4;
  const __bf16* p = base + (col0 + n) * ld + k0 + half * 8;
  bf16x8 lo = *(const bf16x8*)p;
  bf16x8 hi = *(const bf16x8*)(p + 16);
  bf16x16 r;
#pragma unroll
  for (int i = 0; i < 8; ++i) { r[i] = lo[i]; r[i + 8] = hi[i]; }
  return r;
}

// ---------------------------------------------------------------------------
// Kernel 1: fp32 -> bf16 staging of x, W_qkv, W_out
// ---------------------------------------------------------------------------
__global__ void cvt_bf16_kernel(const float* __restrict__ x,
                                const float* __restrict__ wqkv,
                                const float* __restrict__ wout,
                                __bf16* __restrict__ xb,
                                __bf16* __restrict__ wqkvb,
                                __bf16* __restrict__ woutb) {
  int stride = gridDim.x * blockDim.x;
  int i0 = blockIdx.x * blockDim.x + threadIdx.x;
  for (int i = i0; i < T_SEQ * C_DIM; i += stride) xb[i]    = (__bf16)x[i];
  for (int i = i0; i < QKV_N * C_DIM; i += stride) wqkvb[i] = (__bf16)wqkv[i];
  for (int i = i0; i < C_DIM * C_DIM; i += stride) woutb[i] = (__bf16)wout[i];
}

// ---------------------------------------------------------------------------
// Shared GEMM body: one wave computes a 32(M) x 64(N) tile of A[M][K] @ B[N][K]^T.
// All 6 operand tiles of a k-step load before the 8 WMMAs -> loads clause behind
// one s_wait, WMMAs issue back-to-back.
// acc[mt*4+nt] covers rows m0+mt*16, cols n0+nt*16.
// ---------------------------------------------------------------------------
__device__ __forceinline__ void gemm32x64(const __bf16* __restrict__ A,
                                          const __bf16* __restrict__ B,
                                          int m0, int n0, int K, f32x8 acc[8]) {
  for (int k0 = 0; k0 < K; k0 += 32) {
    if (k0 + 64 < K) {
      __builtin_prefetch(A + m0 * K + k0 + 64, 0, 1);
      __builtin_prefetch(B + n0 * K + k0 + 64, 0, 1);
    }
    bf16x16 a0 = load_a16x32(A, K, m0, k0);
    bf16x16 a1 = load_a16x32(A, K, m0 + 16, k0);
    bf16x16 b0 = load_bT16x32(B, K, n0,      k0);
    bf16x16 b1 = load_bT16x32(B, K, n0 + 16, k0);
    bf16x16 b2 = load_bT16x32(B, K, n0 + 32, k0);
    bf16x16 b3 = load_bT16x32(B, K, n0 + 48, k0);
    acc[0] = WMMA_BF16(a0, b0, acc[0]);
    acc[1] = WMMA_BF16(a0, b1, acc[1]);
    acc[2] = WMMA_BF16(a0, b2, acc[2]);
    acc[3] = WMMA_BF16(a0, b3, acc[3]);
    acc[4] = WMMA_BF16(a1, b0, acc[4]);
    acc[5] = WMMA_BF16(a1, b1, acc[5]);
    acc[6] = WMMA_BF16(a1, b2, acc[6]);
    acc[7] = WMMA_BF16(a1, b3, acc[7]);
  }
}

// ---------------------------------------------------------------------------
// Kernel 2: QKV GEMM  qkv[t][n] = sum_k xb[t][k] * Wqkv[n][k]
// Epilogue scatters into head-major Q[H][T][D] (pre-scaled by 0.125),
// K[H][T][D], Vt[H][D][T] (bf16).
// ---------------------------------------------------------------------------
__global__ __launch_bounds__(256) void qkv_gemm_kernel(
    const __bf16* __restrict__ Xb, const __bf16* __restrict__ Wb,
    __bf16* __restrict__ Q, __bf16* __restrict__ Km, __bf16* __restrict__ Vt) {
  int wave = threadIdx.x >> 5;
  int m0 = blockIdx.y * 256 + wave * 32;
  int n0 = blockIdx.x * 64;
  f32x8 acc[8] = {};
  gemm32x64(Xb, Wb, m0, n0, C_DIM, acc);

  int lane = threadIdx.x & 31;
  int nh = lane & 15, half = lane >> 4;
#pragma unroll
  for (int mt = 0; mt < 2; ++mt) {
    int t0 = m0 + mt * 16 + half * 8;     // C layout: row = r + 8*half
#pragma unroll
    for (int nt = 0; nt < 4; ++nt) {
      f32x8 c = acc[mt * 4 + nt];
      int n = n0 + nt * 16 + nh;
      int which = n / C_DIM;              // 0=Q 1=K 2=V
      int cn = n % C_DIM;
      int head = cn / HD, d = cn % HD;
      if (which == 2) {                   // V stored transposed: Vt[head][d][t]
        __bf16* p = Vt + (head * HD + d) * T_SEQ + t0;
#pragma unroll
        for (int r = 0; r < 8; ++r) p[r] = (__bf16)c[r];
      } else {                            // Q/K row-major per head: [head][t][d]
        float s = (which == 0) ? 0.125f : 1.0f;   // fold 1/sqrt(D) into Q (exact)
        __bf16* p = (which == 0 ? Q : Km) + (head * T_SEQ + t0) * HD + d;
#pragma unroll
        for (int r = 0; r < 8; ++r) p[r * HD] = (__bf16)(c[r] * s);
      }
    }
  }
}

// ---------------------------------------------------------------------------
// Kernel 3: flash attention (one head x 64 query rows per block, 4 waves).
// Wave = 16 query rows x full D=64. 32 keys per iteration.
// Row-sum l is accumulated by an extra WMMA against an all-ones B operand
// (same alpha-rescaled recurrence as the output accumulators).
// ---------------------------------------------------------------------------
template <bool MASK>
__device__ __forceinline__ void attn_step(
    int kb, int q0, int wave, int lane,
    const __bf16* __restrict__ Kh, const __bf16* __restrict__ Vh,
    const bf16x16& qa0, const bf16x16& qa1, const bf16x16& ones,
    __bf16 (*pbuf)[32], f32x8 acc[4], f32x8& accl, float mrow[8]) {
  int nh = lane & 15, half = lane >> 4;
  // ---- S = Q K^T for 32 keys (4 WMMAs, k-dim = d = 64; Q pre-scaled) ----
  f32x8 s0 = {}, s1 = {};
  {
    bf16x16 b;
    b = load_bT16x32(Kh, HD, kb,      0);  s0 = WMMA_BF16(qa0, b, s0);
    b = load_bT16x32(Kh, HD, kb,     32);  s0 = WMMA_BF16(qa1, b, s0);
    b = load_bT16x32(Kh, HD, kb + 16, 0);  s1 = WMMA_BF16(qa0, b, s1);
    b = load_bT16x32(Kh, HD, kb + 16, 32); s1 = WMMA_BF16(qa1, b, s1);
  }
  int col0 = kb + nh, col1 = kb + 16 + nh;
#pragma unroll
  for (int r = 0; r < 8; ++r) {
    float v0 = s0[r], v1 = s1[r];
    if (MASK) {
      int row = q0 + r + 8 * half;
      if (col0 > row) v0 = -1e30f;
      if (col1 > row) v1 = -1e30f;
    }
    float mx = fmaxf(v0, v1);
#pragma unroll
    for (int off = 8; off; off >>= 1) mx = fmaxf(mx, __shfl_xor(mx, off, 32));
    float mnew  = fmaxf(mrow[r], mx);
    float alpha = __expf(mrow[r] - mnew);
    mrow[r] = mnew;
    float p0 = __expf(v0 - mnew);
    float p1 = __expf(v1 - mnew);
    acc[0][r] *= alpha; acc[1][r] *= alpha;
    acc[2][r] *= alpha; acc[3][r] *= alpha;
    accl[r]   *= alpha;
    // stash P (C-layout -> row-major LDS tile) for A-layout reload
    pbuf[r + 8 * half][nh]      = (__bf16)p0;
    pbuf[r + 8 * half][16 + nh] = (__bf16)p1;
  }
  // cross-lane LDS RAW inside one wave: compiler can't see it -> wait DScnt
  asm volatile("s_wait_dscnt 0" ::: "memory");
  bf16x16 pa = load_a16x32(&pbuf[0][0], 32, 0, 0);
  // ---- acc += P V ; accl += P 1  (5 WMMAs, k-dim = 32 keys) ----
#pragma unroll
  for (int dt = 0; dt < 4; ++dt) {
    bf16x16 vb = load_bT16x32(Vh, T_SEQ, dt * 16, kb);
    acc[dt] = WMMA_BF16(pa, vb, acc[dt]);
  }
  accl = WMMA_BF16(pa, ones, accl);
}

__global__ __launch_bounds__(128) void attn_kernel(
    const __bf16* __restrict__ Q, const __bf16* __restrict__ Km,
    const __bf16* __restrict__ Vt, __bf16* __restrict__ Ob) {
  __shared__ __bf16 pstage[4][16][32];        // per-wave P tile staging (1KB each)
  int wave = threadIdx.x >> 5;
  int lane = threadIdx.x & 31;
  int nh = lane & 15, half = lane >> 4;
  int head = blockIdx.y;
  // reversed order: longest causal rows launch first (better tail utilization)
  int bx = gridDim.x - 1 - blockIdx.x;
  int q0 = bx * 64 + wave * 16;               // this wave's query-row base

  const __bf16* Qh = Q  + head * T_SEQ * HD;
  const __bf16* Kh = Km + head * T_SEQ * HD;
  const __bf16* Vh = Vt + head * HD * T_SEQ;

  // Q tile stays resident: 16 rows x 64 d  ->  two A operands
  bf16x16 qa0 = load_a16x32(Qh, HD, q0, 0);
  bf16x16 qa1 = load_a16x32(Qh, HD, q0, 32);
  bf16x16 ones;
#pragma unroll
  for (int i = 0; i < 16; ++i) ones[i] = (__bf16)1.0f;

  f32x8 acc[4] = {};
  f32x8 accl = {};
  float mrow[8];
#pragma unroll
  for (int r = 0; r < 8; ++r) mrow[r] = -1e30f;

  int nfull = (q0 + 1) >> 5;                  // blocks with kb+31 <= q0: mask-free
  for (int j = 0; j < nfull; ++j)
    attn_step<false>(j * 32, q0, wave, lane, Kh, Vh, qa0, qa1, ones,
                     pstage[wave], acc, accl, mrow);
  int kb = nfull * 32;                        // single diagonal (masked) block
  if (kb < q0 + 16)
    attn_step<true>(kb, q0, wave, lane, Kh, Vh, qa0, qa1, ones,
                    pstage[wave], acc, accl, mrow);

  // ---- epilogue: normalize rows by l (= any column of accl), merge heads ----
#pragma unroll
  for (int dt = 0; dt < 4; ++dt) {
#pragma unroll
    for (int r = 0; r < 8; ++r) {
      int row = q0 + r + 8 * half;
      float o = acc[dt][r] / accl[r];
      Ob[row * C_DIM + head * HD + dt * 16 + nh] = (__bf16)o;
    }
  }
}

// ---------------------------------------------------------------------------
// Kernel 4: output GEMM  out[t][n] = sum_k Ob[t][k] * Wout[n][k]  (f32 out)
// ---------------------------------------------------------------------------
__global__ __launch_bounds__(256) void out_gemm_kernel(
    const __bf16* __restrict__ Ob, const __bf16* __restrict__ Wob,
    float* __restrict__ out) {
  int wave = threadIdx.x >> 5;
  int m0 = blockIdx.y * 256 + wave * 32;
  int n0 = blockIdx.x * 64;
  f32x8 acc[8] = {};
  gemm32x64(Ob, Wob, m0, n0, C_DIM, acc);

  int lane = threadIdx.x & 31;
  int nh = lane & 15, half = lane >> 4;
#pragma unroll
  for (int mt = 0; mt < 2; ++mt)
#pragma unroll
    for (int nt = 0; nt < 4; ++nt)
#pragma unroll
      for (int r = 0; r < 8; ++r)
        out[(m0 + mt * 16 + half * 8 + r) * C_DIM + n0 + nt * 16 + nh] =
            acc[mt * 4 + nt][r];
}

// ---------------------------------------------------------------------------
extern "C" void kernel_launch(void* const* d_in, const int* in_sizes, int n_in,
                              void* d_out, int out_size, void* d_ws, size_t ws_size,
                              hipStream_t stream) {
  const float* x    = (const float*)d_in[0];   // [1,4096,768]
  const float* wqkv = (const float*)d_in[1];   // [2304,768]
  const float* wout = (const float*)d_in[2];   // [768,768]
  float* out = (float*)d_out;                  // [1,4096,768]

  // workspace partition (bytes; all 256-aligned), total ~36 MB (L2-resident)
  char* ws = (char*)d_ws;
  size_t off = 0;
  __bf16* Xb    = (__bf16*)(ws + off); off += (size_t)T_SEQ * C_DIM * 2;
  __bf16* Wqkvb = (__bf16*)(ws + off); off += (size_t)QKV_N * C_DIM * 2;
  __bf16* Woutb = (__bf16*)(ws + off); off += (size_t)C_DIM * C_DIM * 2;
  __bf16* Q     = (__bf16*)(ws + off); off += (size_t)NH * T_SEQ * HD * 2;
  __bf16* Km    = (__bf16*)(ws + off); off += (size_t)NH * T_SEQ * HD * 2;
  __bf16* Vt    = (__bf16*)(ws + off); off += (size_t)NH * HD * T_SEQ * 2;
  __bf16* Ob    = (__bf16*)(ws + off); off += (size_t)T_SEQ * C_DIM * 2;
  (void)ws_size; (void)in_sizes; (void)n_in; (void)out_size;

  cvt_bf16_kernel<<<1024, 256, 0, stream>>>(x, wqkv, wout, Xb, Wqkvb, Woutb);

  qkv_gemm_kernel<<<dim3(QKV_N / 64, T_SEQ / 256), 256, 0, stream>>>(
      Xb, Wqkvb, Q, Km, Vt);

  attn_kernel<<<dim3(T_SEQ / 64, NH), 128, 0, stream>>>(Q, Km, Vt, Ob);

  out_gemm_kernel<<<dim3(C_DIM / 64, T_SEQ / 256), 256, 0, stream>>>(
      Ob, Woutb, out);
}
